// ProtocolTreeGAttention_44770739094120
// MI455X (gfx1250) — compile-verified, matrix operand order checked by reading, not run on hardware
//
#include <hip/hip_runtime.h>

// ---------------- types ----------------
typedef __attribute__((ext_vector_type(16))) _Float16 v16h;
typedef __attribute__((ext_vector_type(8)))  _Float16 v8h;
typedef __attribute__((ext_vector_type(8)))  float    v8f;
typedef __attribute__((ext_vector_type(4)))  float    v4f;

// ---------------- problem constants ----------------
#define NB    2048
#define FF    32
#define NN    (NB*FF)        // 65536 nodes
#define DIN   64
#define HID   128
#define HH    4
#define EE_RAW 262144
#define EE_TOT (EE_RAW + NN) // edges + self loops = 327680
#define NC    10

#define KMAX  512
#define KPAD  4              // f32 pad per LDS column: stride K+4 dwords == 4 mod 64 banks

// convert 8 consecutive floats (32B aligned) to v8h via two b128 loads
__device__ inline v8h cvt8(const float* __restrict__ p)
{
    const v4f* q = reinterpret_cast<const v4f*>(p);
    v4f u0 = q[0], u1 = q[1];
    v8h r;
    #pragma unroll
    for (int i = 0; i < 4; ++i) { r[i] = (_Float16)u0[i]; r[4 + i] = (_Float16)u1[i]; }
    return r;
}

// B fragment from LDS: 16 f32 (K offsets kb+{0..7,16..23}) -> v16h, 4x ds_load_b128
__device__ inline v16h lds_frag16(const float* lb)
{
    v4f b0 = *reinterpret_cast<const v4f*>(lb);
    v4f b1 = *reinterpret_cast<const v4f*>(lb + 4);
    v4f b2 = *reinterpret_cast<const v4f*>(lb + 16);
    v4f b3 = *reinterpret_cast<const v4f*>(lb + 20);
    v16h r;
    #pragma unroll
    for (int i = 0; i < 4; ++i) {
        r[i]      = (_Float16)b0[i];
        r[4 + i]  = (_Float16)b1[i];
        r[8 + i]  = (_Float16)b2[i];
        r[12 + i] = (_Float16)b3[i];
    }
    return r;
}

// =====================================================================
// WMMA GEMM: C[M,Nc] = act(A[M,K] @ B[K,Nc] + bias)
// fp32 in/out, f16 WMMA 16x16x32, fp32 accumulate.
// Block = 256 threads = 8 waves. The block's 16-column B panel is DMA'd
// global->LDS with per-lane GLOBAL_LOAD_ASYNC_TO_LDS_B32 (ASYNCcnt), then
// each wave owns a 16x16 output tile: rows blockIdx.y*128+wave*16,
// cols blockIdx.x*16. Requires K%32==0, M%128==0, Nc%16==0, 32B-aligned rows.
// ACT: 0 = none, 1 = leaky_relu(0.01)
// =====================================================================
template<int ACT>
__global__ __launch_bounds__(256)
void wmma_gemm(const float* __restrict__ A, const float* __restrict__ Bm,
               const float* __restrict__ bias, float* __restrict__ C,
               int M, int K, int Ncols)
{
    __shared__ float ldsB[16 * (KMAX + KPAD)];   // 33 KB, column-major f32, padded

    const int lane = threadIdx.x & 31;
    const int wave = threadIdx.x >> 5;
    const int rowBase = blockIdx.y * 128 + wave * 16;
    const int colBase = blockIdx.x * 16;
    const int l15 = lane & 15;
    const int hi  = lane >> 4;          // 0: K chunk {0..7,16..23}; 1: {8..15,24..31}
    const int kb  = hi ? 8 : 0;
    const int ldc = K + KPAD;           // LDS column stride (f32 elements)

    // ---- async DMA B panel into LDS: cols [colBase,colBase+16), all K ----
    // thread t copies elements (k = t>>4 + 16*i, c = t&15): the 16 lanes of a
    // row segment read 64B coalesced; each lane scatters to its own LDS addr.
    {
        const int c = threadIdx.x & 15;
        for (int k = threadIdx.x >> 4; k < K; k += 16) {
            unsigned    lo = (unsigned)(uintptr_t)&ldsB[c * ldc + k];
            const float* gp = Bm + (size_t)k * Ncols + colBase + c;
            asm volatile("global_load_async_to_lds_b32 %0, %1, off"
                         :: "v"(lo), "v"(gp) : "memory");
        }
    }
    asm volatile("s_wait_asynccnt 0" ::: "memory");
    __syncthreads();

    const int arow = rowBase + l15;     // A row held by this lane (ISA A layout)
    const int col  = colBase + l15;     // B/C column held by this lane
    const float* abase = A + (size_t)arow * K + kb;
    const float* bbase = &ldsB[l15 * ldc + kb];

    v8f acc = {};
    for (int k0 = 0; k0 < K; k0 += 32) {
        const float* ap = abase + k0;
        if (k0 + 32 < K) __builtin_prefetch(ap + 32, 0, 3);   // next A fragment
        // A fragment: 2 x 8 consecutive f32 -> f16  (global_load_b128 x4)
        v8h a0 = cvt8(ap);
        v8h a1 = cvt8(ap + 16);
        v16h a = __builtin_shufflevector(a0, a1, 0,1,2,3,4,5,6,7,8,9,10,11,12,13,14,15);
        // B fragment: 4 x 16B aligned ds_load_b128 + in-register cvt
        v16h b = lds_frag16(bbase + k0);
        acc = __builtin_amdgcn_wmma_f32_16x16x32_f16(
                  /*neg_a=*/false, a, /*neg_b=*/false, b,
                  /*c_mod=*/(short)0, acc, /*reuse_a=*/false, /*reuse_b=*/false);
    }

    const float bv = bias ? bias[col] : 0.0f;
    #pragma unroll
    for (int v = 0; v < 8; ++v) {
        float val = acc[v] + bv;
        if (ACT == 1) val = (val > 0.0f) ? val : 0.01f * val;
        const int row = rowBase + v + (hi ? 8 : 0);   // ISA C/D layout
        C[(size_t)row * Ncols + col] = val;           // per-row contiguous across lanes
    }
}

// =====================================================================
// Elementwise / helper kernels
// =====================================================================
__global__ void fill_f32(float* p, size_t n, float v)
{
    size_t t = (size_t)blockIdx.x * blockDim.x + threadIdx.x;
    size_t stride = (size_t)gridDim.x * blockDim.x;
    for (; t < n; t += stride) p[t] = v;
}

// per-(node,head) attention logits: als/ald[n*H+h] = dot(hx[n,h,:], att_{s,d}[h,:])
__global__ void att_logits(const float* __restrict__ hx,
                           const float* __restrict__ att_s,
                           const float* __restrict__ att_d,
                           float* __restrict__ als, float* __restrict__ ald,
                           int n, int Hh, int C)
{
    int t = blockIdx.x * blockDim.x + threadIdx.x;
    int total = n * Hh;
    if (t >= total) return;
    int node = t / Hh, h = t - node * Hh;
    const float* p = hx + (size_t)node * Hh * C + (size_t)h * C;
    float ss = 0.0f, sd = 0.0f;
    for (int c = 0; c < C; ++c) {
        float v = p[c];
        ss += v * att_s[h * C + c];
        sd += v * att_d[h * C + c];
    }
    als[t] = ss;
    ald[t] = sd;
}

__device__ inline void edge_sd(const int* __restrict__ ei, int e, int& s, int& d)
{
    if (e < EE_RAW) { s = ei[e]; d = ei[EE_RAW + e]; }
    else            { s = e - EE_RAW; d = s; }        // self loop
}

__device__ inline void atomicMaxFloat(float* addr, float val)
{
    unsigned int* a = (unsigned int*)addr;
    unsigned int old = *a;
    while (true) {
        float cur = __uint_as_float(old);
        if (cur >= val) break;
        unsigned int assumed = old;
        old = atomicCAS(a, assumed, __float_as_uint(val));
        if (old == assumed) break;
    }
}

// pass 1: segment max over dst of leaky_relu(als[src]+ald[dst], 0.2)
__global__ void edge_max(const int* __restrict__ ei,
                         const float* __restrict__ als, const float* __restrict__ ald,
                         float* __restrict__ m, int Hh)
{
    int total = EE_TOT * Hh;
    int stride = gridDim.x * blockDim.x;
    for (int t = blockIdx.x * blockDim.x + threadIdx.x; t < total; t += stride) {
        int e = t / Hh, h = t - e * Hh;
        int s, d; edge_sd(ei, e, s, d);
        float v = als[s * Hh + h] + ald[d * Hh + h];
        v = (v > 0.0f) ? v : 0.2f * v;
        atomicMaxFloat(&m[d * Hh + h], v);
    }
}

// pass 2: ex = exp(e - m[dst]); denom[dst] += ex
__global__ void edge_expsum(const int* __restrict__ ei,
                            const float* __restrict__ als, const float* __restrict__ ald,
                            const float* __restrict__ m,
                            float* __restrict__ ex, float* __restrict__ den, int Hh)
{
    int total = EE_TOT * Hh;
    int stride = gridDim.x * blockDim.x;
    for (int t = blockIdx.x * blockDim.x + threadIdx.x; t < total; t += stride) {
        int e = t / Hh, h = t - e * Hh;
        int s, d; edge_sd(ei, e, s, d);
        float v = als[s * Hh + h] + ald[d * Hh + h];
        v = (v > 0.0f) ? v : 0.2f * v;
        float mc = m[d * Hh + h];
        mc = (mc < -1e29f) ? 0.0f : mc;          // isfinite() fixup from reference
        float xv = __expf(v - mc);
        ex[t] = xv;
        atomicAdd(&den[d * Hh + h], xv);
    }
}

// pass 3: out[dst,h,:] += (ex/(den[dst,h]+eps)) * hx[src,h,:]
__global__ void edge_agg(const int* __restrict__ ei,
                         const float* __restrict__ ex, const float* __restrict__ den,
                         const float* __restrict__ hx, float* __restrict__ out,
                         int Hh, int C)
{
    const int HC = Hh * C;
    for (int e = blockIdx.x; e < EE_TOT; e += gridDim.x) {
        int s, d; edge_sd(ei, e, s, d);
        for (int idx = threadIdx.x; idx < HC; idx += blockDim.x) {
            int h = idx / C;
            float alpha = ex[e * Hh + h] / (den[d * Hh + h] + 1e-16f);
            atomicAdd(&out[(size_t)d * HC + idx], alpha * hx[(size_t)s * HC + idx]);
        }
    }
}

// mode 0: p += bias ; mode 1: p = elu(p + bias)
__global__ void bias_act(float* __restrict__ p, const float* __restrict__ bias,
                         size_t n, int C, int mode)
{
    size_t stride = (size_t)gridDim.x * blockDim.x;
    for (size_t t = (size_t)blockIdx.x * blockDim.x + threadIdx.x; t < n; t += stride) {
        float v = p[t] + bias[t % C];
        if (mode == 1) v = (v > 0.0f) ? v : (__expf(v) - 1.0f);
        p[t] = v;
    }
}

// per-graph mean over the F=32 consecutive nodes (batch = node // 32)
__global__ void pool_mean(const float* __restrict__ x, float* __restrict__ g)
{
    int b = blockIdx.x;       // 0..NB-1
    int c = threadIdx.x;      // 0..HID-1
    float s = 0.0f;
    for (int f = 0; f < FF; ++f)
        s += x[((size_t)b * FF + f) * HID + c];
    g[(size_t)b * HID + c] = s * (1.0f / (float)FF);
}

// final tiny GEMM: out[2048,10] = hcls[2048,64] @ Wc2[64,10] + bc2
__global__ void head_gemm(const float* __restrict__ hcls, const float* __restrict__ Wc2,
                          const float* __restrict__ bc2, float* __restrict__ out)
{
    int t = blockIdx.x * blockDim.x + threadIdx.x;
    if (t >= NB * NC) return;
    int r = t / NC, c = t - r * NC;
    float s = bc2[c];
    for (int k = 0; k < 64; ++k)
        s += hcls[(size_t)r * 64 + k] * Wc2[k * NC + c];
    out[t] = s;
}

// =====================================================================
// host launch
// =====================================================================
extern "C" void kernel_launch(void* const* d_in, const int* in_sizes, int n_in,
                              void* d_out, int out_size, void* d_ws, size_t ws_size,
                              hipStream_t stream)
{
    (void)in_sizes; (void)n_in; (void)out_size; (void)ws_size;

    const float* emb      = (const float*)d_in[0];
    const int*   ei       = (const int*)  d_in[1];
    /* d_in[2] = batch (implicit: node // 32) */
    const float* W_align  = (const float*)d_in[3];
    const float* b_align  = (const float*)d_in[4];
    const float* W1       = (const float*)d_in[5];
    const float* att_src1 = (const float*)d_in[6];
    const float* att_dst1 = (const float*)d_in[7];
    const float* b1       = (const float*)d_in[8];
    const float* W2       = (const float*)d_in[9];
    const float* att_src2 = (const float*)d_in[10];
    const float* att_dst2 = (const float*)d_in[11];
    const float* b2       = (const float*)d_in[12];
    const float* Wc1      = (const float*)d_in[13];
    const float* bc1      = (const float*)d_in[14];
    const float* Wc2      = (const float*)d_in[15];
    const float* bc2      = (const float*)d_in[16];
    float* out = (float*)d_out;
    float* ws  = (float*)d_ws;

    // ---- workspace layout (floats) ----
    size_t off = 0;
    float* x     = ws + off; off += (size_t)NN * HID;        // 8.4M  (reused for hx2)
    float* hx1   = ws + off; off += (size_t)NN * (HH * HID); // 33.6M
    float* out1  = ws + off; off += (size_t)NN * (HH * HID); // 33.6M
    float* als1  = ws + off; off += (size_t)NN * HH;
    float* ald1  = ws + off; off += (size_t)NN * HH;
    float* m1    = ws + off; off += (size_t)NN * HH;
    float* den1  = ws + off; off += (size_t)NN * HH;
    float* ex1   = ws + off; off += (size_t)EE_TOT * HH;
    float* out2  = ws + off; off += (size_t)NN * HID;
    float* als2  = ws + off; off += (size_t)NN;
    float* ald2  = ws + off; off += (size_t)NN;
    float* m2    = ws + off; off += (size_t)NN;
    float* den2  = ws + off; off += (size_t)NN;
    float* ex2   = ws + off; off += (size_t)EE_TOT;
    float* g     = ws + off; off += (size_t)NB * HID;
    float* hcls  = ws + off; off += (size_t)NB * 64;
    float* hx2   = x;   // x is dead after the GAT1 transform

    // 1) align: x = emb @ W_align + b_align   [65536,64]x[64,128]
    wmma_gemm<0><<<dim3(HID / 16, NN / 128), 256, 0, stream>>>(
        emb, W_align, b_align, x, NN, DIN, HID);

    // 2) GAT1 transform: hx1 = x @ W1   [65536,128]x[128,512]
    wmma_gemm<0><<<dim3((HH * HID) / 16, NN / 128), 256, 0, stream>>>(
        x, W1, nullptr, hx1, NN, HID, HH * HID);

    // 3) attention logits (H=4)
    att_logits<<<(NN * HH + 255) / 256, 256, 0, stream>>>(
        hx1, att_src1, att_dst1, als1, ald1, NN, HH, HID);

    // 4) init segment buffers
    fill_f32<<<1024, 256, 0, stream>>>(m1,   (size_t)NN * HH, -1e30f);
    fill_f32<<<1024, 256, 0, stream>>>(den1, (size_t)NN * HH, 0.0f);
    fill_f32<<<4096, 256, 0, stream>>>(out1, (size_t)NN * HH * HID, 0.0f);

    // 5-7) edge softmax + aggregate (GAT1)
    edge_max   <<<4096, 256, 0, stream>>>(ei, als1, ald1, m1, HH);
    edge_expsum<<<4096, 256, 0, stream>>>(ei, als1, ald1, m1, ex1, den1, HH);
    edge_agg   <<<16384, 256, 0, stream>>>(ei, ex1, den1, hx1, out1, HH, HID);

    // 8) out1 = elu(out1 + b1)
    bias_act<<<4096, 256, 0, stream>>>(out1, b1, (size_t)NN * HH * HID, HH * HID, 1);

    // 9) GAT2 transform: hx2 = out1 @ W2   [65536,512]x[512,128]
    wmma_gemm<0><<<dim3(HID / 16, NN / 128), 256, 0, stream>>>(
        out1, W2, nullptr, hx2, NN, HH * HID, HID);

    // 10) attention logits (H=1)
    att_logits<<<(NN + 255) / 256, 256, 0, stream>>>(
        hx2, att_src2, att_dst2, als2, ald2, NN, 1, HID);

    // 11) init
    fill_f32<<<512,  256, 0, stream>>>(m2,   (size_t)NN, -1e30f);
    fill_f32<<<512,  256, 0, stream>>>(den2, (size_t)NN, 0.0f);
    fill_f32<<<2048, 256, 0, stream>>>(out2, (size_t)NN * HID, 0.0f);

    // 12) edge softmax + aggregate (GAT2)
    edge_max   <<<2048, 256, 0, stream>>>(ei, als2, ald2, m2, 1);
    edge_expsum<<<2048, 256, 0, stream>>>(ei, als2, ald2, m2, ex2, den2, 1);
    edge_agg   <<<16384, 128, 0, stream>>>(ei, ex2, den2, hx2, out2, 1, HID);

    // 13) out2 += b2
    bias_act<<<2048, 256, 0, stream>>>(out2, b2, (size_t)NN * HID, HID, 0);

    // 14) global mean pool -> g [2048,128]
    pool_mean<<<NB, HID, 0, stream>>>(out2, g);

    // 15) classifier layer 1 (WMMA): hcls = leaky_relu(g @ Wc1 + bc1, 0.01)
    wmma_gemm<1><<<dim3(64 / 16, NB / 128), 256, 0, stream>>>(
        g, Wc1, bc1, hcls, NB, HID, 64);

    // 16) classifier layer 2 -> out [2048,10]
    head_gemm<<<(NB * NC + 255) / 256, 256, 0, stream>>>(hcls, Wc2, bc2, out);
}
